// TGA_UNet_18949395710254
// MI455X (gfx1250) — matrix operand order repaired and checked by reading, not compiled, not run on hardware
//
#include <hip/hip_runtime.h>
#include <hip/hip_bf16.h>
#include <math.h>

// ============================================================================
// TGA U-Net for MI455X (gfx1250), compile-only build.
//
// Dense GEMMs: V_WMMA_F32_16X16X32_BF16, fp32 accumulate. Weights are
// pre-packed once per layer into bf16 *WMMA-B fragment order* so each
// 32(K)x16(N) tile is a contiguous 1KB blob; the GEMM inner loop then
// double-buffers B tiles in LDS via the Tensor Data Mover
// (tensor_load_to_lds + s_wait_tensorcnt), overlapping the DMA with the
// WMMA on the current tile. A-operand: 4x b128 fp32 loads + cvt_pk_bf16.
// All GEMM shapes have M%16==0, N%16==0, K%64==0 (no remainders, EXEC all-1s).
// Graph ops (edge softmax, segment sums, top-k, scatter/gather) are scalar
// bandwidth/atomic-bound kernels. Top-k is a 4-pass radix select on device.
// ============================================================================

#define HEADS 4
#define N0    100000
#define EDGES 1600000
#define KK1   72000
#define KK2   40320
#define KK3   16128

typedef __bf16 bf16x16 __attribute__((ext_vector_type(16)));
typedef float  f32x8   __attribute__((ext_vector_type(8)));
typedef unsigned int u32x4 __attribute__((ext_vector_type(4)));
typedef int          i32x8 __attribute__((ext_vector_type(8)));
typedef int          i32x4 __attribute__((ext_vector_type(4)));

#ifndef __has_builtin
#define __has_builtin(x) 0
#endif
#if __has_builtin(__builtin_amdgcn_tensor_load_to_lds) && \
    __has_builtin(__builtin_amdgcn_s_wait_tensorcnt)
#define TGA_HAVE_TDM 1
#else
#define TGA_HAVE_TDM 0
#endif

#define GSTRIDE(i, n) for (size_t i = (size_t)blockIdx.x * blockDim.x + threadIdx.x; \
                           i < (size_t)(n); i += (size_t)gridDim.x * blockDim.x)

// ------------------------------ device helpers ------------------------------

__device__ __forceinline__ unsigned f2mono(float f) {
  unsigned u = __float_as_uint(f);
  return (u >> 31) ? ~u : (u | 0x80000000u);
}
__device__ __forceinline__ float mono2f(unsigned u) {
  return __uint_as_float((u >> 31) ? (u & 0x7FFFFFFFu) : ~u);
}

#if TGA_HAVE_TDM
// 1-D TDM copy of one packed B tile (512 bf16 = 1KB, contiguous) into LDS.
// D# group0: count=1 | lds_addr | global_addr[31:0] | addr[56:32] + type=2.
// D# group1: data_size=2B; tensor_dim0=2^30 (no OOB clamp); tensor_dim1=1;
//            tile_dim0=512; tile_dim1/2 unused; dim0_stride=2^30.
__device__ __forceinline__ void tdm_issue(const __bf16* sh, const __bf16* gsrc) {
  unsigned long long ga = (unsigned long long)(size_t)gsrc;
  u32x4 g0 = { 1u, (unsigned)(size_t)sh, (unsigned)ga,
               (unsigned)((ga >> 32) & 0x01FFFFFFu) | 0x80000000u };
  i32x8 g1 = { 0x00010000,            // workgroup_mask=0, data_size=1 (2B)
               0,                     // atomic barrier addr = 0, dim0[15:0]=0
               0x00014000,            // dim0[31:16]=0x4000 (2^30), dim1[15:0]=1
               0x02000000,            // dim1[31:16]=0, tile_dim0=512
               0,                     // tile_dim1=0, tile_dim2=0 (unused)
               0x40000000,            // dim0_stride[31:0] = 2^30
               0, 0 };                // stride hi bits, dim1_stride = 0
  i32x4 z4 = { 0, 0, 0, 0 };
#if defined(__clang_major__) && __clang_major__ >= 23
  i32x8 z8 = { 0, 0, 0, 0, 0, 0, 0, 0 };
  __builtin_amdgcn_tensor_load_to_lds(g0, g1, z4, z4, z8, 0);
#else
  __builtin_amdgcn_tensor_load_to_lds(g0, g1, z4, z4, 0);
#endif
}
#define BTILE_LOAD(sh, g, ln) tdm_issue((sh), (g))
#define BTILE_WAIT_PREV()     __builtin_amdgcn_s_wait_tensorcnt(1)
#define BTILE_WAIT_ALL()      __builtin_amdgcn_s_wait_tensorcnt(0)
#else
// Fallback: each lane copies its own 32B fragment (lane reads only what it
// wrote, so no synchronization is needed).
#define BTILE_LOAD(sh, g, ln) \
  (*(bf16x16*)((sh) + (size_t)(ln) * 16) = *(const bf16x16*)((g) + (size_t)(ln) * 16))
#define BTILE_WAIT_PREV() ((void)0)
#define BTILE_WAIT_ALL()  ((void)0)
#endif

// --------------------------- weight pre-pack kernel -------------------------
// W fp32 [K,N] row-major -> bf16 tiles, layout [tn][tk][lane][e] so that each
// (tk,tn) tile is 1KB contiguous and lane ln's 16 elements are contiguous.
// Fragment mapping (contiguous k order, same as A in k_gemm): k = 16*(ln>>4) + e, n = ln&15.
__global__ void k_pack_w(const float* __restrict__ W, int ldw,
                         __bf16* __restrict__ out, int nK, int nN) {
  GSTRIDE(i, (size_t)nK * nN * 512) {
    unsigned e  = (unsigned)(i & 15);
    unsigned ln = (unsigned)((i >> 4) & 31);
    size_t tile = i >> 9;
    int tk = (int)(tile % nK);
    int tn = (int)(tile / nK);
    int k = tk * 32 + ((ln >> 4) << 4) + (int)e;
    int n = tn * 16 + (int)(ln & 15);
    out[i] = (__bf16)W[(size_t)k * ldw + n];
  }
}

// ------------------------------ GEMM kernel ---------------------------------
// C[M,N] = act(A[M,K] @ W[K,N] + bias), B pre-packed. 8 waves/block, each
// wave owns a 16x16 tile (block = 128 rows x 16 cols). Per-wave LDS double
// buffer for B tiles, filled by TDM and drained by ds_load_b128 pairs.
__global__ __launch_bounds__(256)
void k_gemm(const float* __restrict__ A, int lda,
            const __bf16* __restrict__ Wpk,
            const float* __restrict__ bias,
            float* __restrict__ C, int ldc,
            int M, int K, int act) {
  __shared__ __align__(32) __bf16 shB[8][2][512];
  const int wv  = threadIdx.x >> 5;
  const int ln  = threadIdx.x & 31;
  const int hh  = ln >> 4;
  const int l15 = ln & 15;
  const int m0  = blockIdx.x * 128 + wv * 16;
  if (m0 >= M) return;                       // wave-uniform: EXEC stays all-1s
  const int tn  = blockIdx.y;
  const int col = tn * 16 + l15;
  const int nK  = K >> 5;
  const float* arow = A + (size_t)(m0 + l15) * lda;
  const __bf16* wbase = Wpk + (size_t)tn * nK * 512;
  __bf16* sh0 = &shB[wv][0][0];
  __bf16* sh1 = &shB[wv][1][0];

  BTILE_LOAD(sh0, wbase, ln);                // tile 0 -> buffer 0
  f32x8 acc = {0.f, 0.f, 0.f, 0.f, 0.f, 0.f, 0.f, 0.f};
  for (int tk = 0; tk < nK; ++tk) {
    if (tk + 1 < nK) {                       // issue next tile, then wait <=1:
      BTILE_LOAD((tk & 1) ? sh0 : sh1, wbase + (size_t)(tk + 1) * 512, ln);
      BTILE_WAIT_PREV();                     // in-order => current tile landed
    } else {
      BTILE_WAIT_ALL();
    }
    const __bf16* bt = ((tk & 1) ? sh1 : sh0) + (size_t)ln * 16;
    bf16x16 bv = *(const bf16x16*)bt;        // 2x ds_load_b128

    const int ka = tk * 32 + hh * 16;        // A fragment: k = 16*hh + e
    __builtin_prefetch(arow + ka + 64, 0, 1);
    const float4 a0 = *(const float4*)(arow + ka);
    const float4 a1 = *(const float4*)(arow + ka + 4);
    const float4 a2 = *(const float4*)(arow + ka + 8);
    const float4 a3 = *(const float4*)(arow + ka + 12);
    float af[16] = {a0.x, a0.y, a0.z, a0.w, a1.x, a1.y, a1.z, a1.w,
                    a2.x, a2.y, a2.z, a2.w, a3.x, a3.y, a3.z, a3.w};
    bf16x16 av;
#pragma unroll
    for (int e = 0; e < 16; ++e) av[e] = (__bf16)af[e];
    acc = __builtin_amdgcn_wmma_f32_16x16x32_bf16(false, av, false, bv,
                                                  (short)0, acc, false, false);
  }
  const float bb = bias ? bias[col] : 0.0f;
#pragma unroll
  for (int r = 0; r < 8; ++r) {
    float v = acc[r] + bb;                   // C layout: VGPR r -> row r + 8*hh
    if (act == 1) v = fmaxf(v, 0.0f);
    C[(size_t)(m0 + r + hh * 8) * ldc + col] = v;
  }
}

// ---------------------------- elementwise kernels ---------------------------

__global__ void k_fill_f32(float* p, float v, size_t n)    { GSTRIDE(i, n) p[i] = v; }
__global__ void k_fill_u32(unsigned* p, unsigned v, size_t n) { GSTRIDE(i, n) p[i] = v; }
__global__ void k_fill_i32(int* p, int v, size_t n)        { GSTRIDE(i, n) p[i] = v; }
__global__ void k_mul(const float* a, const float* b, float* o, size_t n) {
  GSTRIDE(i, n) o[i] = a[i] * b[i];
}
__global__ void k_elu(const float* a, float* o, size_t n) {
  GSTRIDE(i, n) { float v = a[i]; o[i] = v > 0.f ? v : (expf(v) - 1.f); }
}
__global__ void k_add_elu(const float* a, const float* b, float* o, size_t n) {
  GSTRIDE(i, n) { float v = a[i] + b[i]; o[i] = v > 0.f ? v : (expf(v) - 1.f); }
}
__global__ void k_copy_cols(const float* in, int ldi, float* out, int ldo,
                            size_t M, int cols) {
  GSTRIDE(i, M * (size_t)cols) {
    size_t r = i / cols; int j = (int)(i - r * cols);
    out[r * (size_t)ldo + j] = in[r * (size_t)ldi + j];
  }
}

// ------------------------------- GAT kernels --------------------------------

__global__ void k_attn_coef(const float* z, const float* al, const float* ar,
                            float* el, float* er, int n, int Hd) {
  GSTRIDE(i, (size_t)n * HEADS) {
    int h = (int)(i & 3);
    const float* zp = z + (i >> 2) * (size_t)(HEADS * Hd) + (size_t)h * Hd;
    float sl = 0.f, sr = 0.f;
    for (int d = 0; d < Hd; ++d) {
      float zv = zp[d];
      sl += zv * al[h * Hd + d];
      sr += zv * ar[h * Hd + d];
    }
    el[i] = sl; er[i] = sr;
  }
}

__global__ void k_edge_e_max(const int* src, const int* dst, const float* el,
                             const float* er, float* e, unsigned* mm, size_t E4) {
  GSTRIDE(i, E4) {
    size_t eid = i >> 2; int h = (int)(i & 3);
    int s = src[eid], d = dst[eid];
    float v = el[(size_t)s * 4 + h] + er[(size_t)d * 4 + h];
    v = v > 0.f ? v : 0.2f * v;                      // leaky_relu(., 0.2)
    e[i] = v;
    atomicMax(&mm[(size_t)d * 4 + h], f2mono(v));
  }
}

__global__ void k_mono_fix(const unsigned* mm, float* mf, size_t n) {
  GSTRIDE(i, n) {
    float v = mono2f(mm[i]);
    mf[i] = (v * 0.0f == 0.0f) ? v : 0.f;            // finite ? v : 0
  }
}

__global__ void k_edge_p_sum(const int* dst, const float* e, const float* mf,
                             float* p, float* ss, size_t E4) {
  GSTRIDE(i, E4) {
    size_t eid = i >> 2; int h = (int)(i & 3); int d = dst[eid];
    float pv = expf(e[i] - mf[(size_t)d * 4 + h]);
    p[i] = pv;
    atomicAdd(&ss[(size_t)d * 4 + h], pv);
  }
}

__global__ void k_edge_agg(const int* src, const int* dst, const float* p,
                           const float* ss, const float* z, float* out, int Hd) {
  int eid = blockIdx.x;
  int s = src[eid], d = dst[eid];
  int Ds = HEADS * Hd;
  for (int f = threadIdx.x; f < Ds; f += blockDim.x) {
    int h = f / Hd;
    float alpha = p[(size_t)eid * 4 + h] / fmaxf(ss[(size_t)d * 4 + h], 1e-9f);
    atomicAdd(&out[(size_t)d * Ds + f], alpha * z[(size_t)s * Ds + f]);
  }
}

// ------------------------------- GCT / pool ---------------------------------

__global__ void k_gct_gate(const float* h, float* o, int M, int C) {
  int row = blockIdx.x * (blockDim.x >> 5) + (threadIdx.x >> 5);
  int ln = threadIdx.x & 31;
  if (row >= M) return;
  const float* hp = h + (size_t)row * C;
  float mx = -INFINITY, sm = 0.f;
  for (int j = ln; j < C; j += 32) { float v = hp[j]; mx = fmaxf(mx, v); sm += v; }
  for (int off = 16; off; off >>= 1) {
    mx = fmaxf(mx, __shfl_down(mx, off));
    sm += __shfl_down(sm, off);
  }
  mx = __shfl(mx, 0); sm = __shfl(sm, 0);
  float g = 1.f / (1.f + expf(-mx)) + 1.f / (1.f + expf(-sm / (float)C));
  float* op = o + (size_t)row * C;
  for (int j = ln; j < C; j += 32) op[j] = g * hp[j];
}

__global__ void k_rowdot(const float* a, const float* w, const float* b,
                         float* sc, int M, int D) {
  int row = blockIdx.x * (blockDim.x >> 5) + (threadIdx.x >> 5);
  int ln = threadIdx.x & 31;
  if (row >= M) return;
  const float* ap = a + (size_t)row * D;
  float s = 0.f;
  for (int j = ln; j < D; j += 32) s += ap[j] * w[j];
  for (int off = 16; off; off >>= 1) s += __shfl_down(s, off);
  if (ln == 0) sc[row] = s + b[0];
}

// --------------------------- top-k (radix select) ---------------------------

struct TopkState { unsigned prefix, kth, cnt, pad; unsigned hist[256]; };

__global__ void k_topk_init(TopkState* st, unsigned K) {
  if (threadIdx.x == 0) { st->prefix = 0; st->kth = K; st->cnt = 0; }
  st->hist[threadIdx.x] = 0;
}
__global__ void k_topk_hist(const float* sc, int n, TopkState* st, int shift) {
  unsigned pre = st->prefix;
  unsigned hm = (shift == 24) ? 0u : (0xFFFFFFFFu << (shift + 8));
  GSTRIDE(i, n) {
    unsigned key = f2mono(sc[i]);
    if ((key & hm) == (pre & hm)) atomicAdd(&st->hist[(key >> shift) & 255u], 1u);
  }
}
__global__ void k_topk_scan(TopkState* st, int shift) {
  if (threadIdx.x == 0) {
    unsigned kth = st->kth, pre = st->prefix;
    int b = 255;
    for (; b > 0; --b) { unsigned c = st->hist[b]; if (c >= kth) break; kth -= c; }
    st->prefix = pre | ((unsigned)b << shift);
    st->kth = kth;
  }
  __syncthreads();
  st->hist[threadIdx.x] = 0;
}
__global__ void k_topk_gt(const float* sc, int n, TopkState* st, int* idx) {
  unsigned th = st->prefix;
  GSTRIDE(i, n) {
    if (f2mono(sc[i]) > th) { unsigned p = atomicAdd(&st->cnt, 1u); idx[p] = (int)i; }
  }
}
__global__ void k_topk_eq(const float* sc, int n, TopkState* st, int* idx, unsigned K) {
  unsigned th = st->prefix;
  GSTRIDE(i, n) {
    if (f2mono(sc[i]) == th) {
      unsigned p = atomicAdd(&st->cnt, 1u);
      if (p < K) idx[p] = (int)i;
    }
  }
}

// --------------------------- pool / unpool helpers --------------------------

__global__ void k_set_map(const int* idx, int* map, int K) {
  GSTRIDE(j, (size_t)K) map[idx[j]] = (int)j;
}
__global__ void k_remap(const int* src, const int* dst, const int* map,
                        int* s2, int* d2, int K, size_t E_) {
  GSTRIDE(i, E_) {
    int ms = map[src[i]], md = map[dst[i]];
    bool ok = (ms >= 0) && (md >= 0);
    s2[i] = ok ? ms : 0;
    d2[i] = ok ? md : K;                    // invalid edges -> dump segment K
  }
}
__global__ void k_gather_rows(const float* in, const int* idx, float* out, int D) {
  int j = blockIdx.x;
  const float* ip = in + (size_t)idx[j] * D;
  float* op = out + (size_t)j * D;
  for (int d = threadIdx.x; d < D; d += blockDim.x) op[d] = ip[d];
}
__global__ void k_scatter_add(const float* x, const int* idx, float* up, int D) {
  int j = blockIdx.x;
  float* op = up + (size_t)idx[j] * D;
  const float* ip = x + (size_t)j * D;
  for (int d = threadIdx.x; d < D; d += blockDim.x) op[d] += ip[d];
}

// -------------------------------- GCN head ----------------------------------

__global__ void k_degree(const int* src, const int* dst, float* ns, float* nd, size_t E_) {
  GSTRIDE(i, E_) { atomicAdd(&ns[src[i]], 1.f); atomicAdd(&nd[dst[i]], 1.f); }
}
__global__ void k_invsqrt(float* v, size_t n) {
  GSTRIDE(i, n) v[i] = rsqrtf(fmaxf(v[i], 1.f));
}
__global__ void k_gcn_agg(const int* src, const int* dst, const float* h,
                          const float* ns, float* agg, int D) {
  int eid = blockIdx.x;
  int s = src[eid], d = dst[eid];
  float w = ns[s];
  const float* hp = h + (size_t)s * D;
  float* op = agg + (size_t)d * D;
  for (int f = threadIdx.x; f < D; f += blockDim.x) atomicAdd(&op[f], w * hp[f]);
}
__global__ void k_scale_rows(float* a, const float* nd, int M, int D) {
  GSTRIDE(i, (size_t)M * D) a[i] *= nd[i / D];
}

// ============================================================================
// Host-side orchestration
// ============================================================================

struct Lin  { const float *w, *b; };
struct Gat  { const float *al0, *al1, *ar0, *ar1, *w0, *w1, *wres; };
struct Gct  { Lin m1a, m1b, m2a, m2b; };
struct FusP { Lin c13, c31, ffa, ffb, mca, mcb; };
struct EncP { Lin atta, attb; FusP fus; Gat gat; Gct gct; Lin pool; };
struct DecP { Lin atta, attb; Gat gat; Gct gct; };

// Input order assumption: jax-style tree flatten of setup_inputs() dict
// (dict keys sorted recursively): dst, params{dec1,dec2,dec3,enc1,enc2,enc3,
// gcn1,gcn2}, src, x.  Within a linear: {b,w}.  Within gat: {al0,al1,ar0,
// ar1,w0,w1,wres}.  Within enc: {atta,attb,fus{c13,c31,ffa,ffb,mca,mcb},
// gat,gct{m1a..m2b},pool}.  Within dec: {atta,attb,gat,gct}.
struct Cur { void* const* a; int i; const float* f() { return (const float*)a[i++]; } };
static Lin  rdLin(Cur& c) { Lin l; l.b = c.f(); l.w = c.f(); return l; }
static Gat  rdGat(Cur& c) { Gat g; g.al0 = c.f(); g.al1 = c.f(); g.ar0 = c.f();
                            g.ar1 = c.f(); g.w0 = c.f(); g.w1 = c.f(); g.wres = c.f(); return g; }
static Gct  rdGct(Cur& c) { Gct g; g.m1a = rdLin(c); g.m1b = rdLin(c);
                            g.m2a = rdLin(c); g.m2b = rdLin(c); return g; }
static FusP rdFus(Cur& c) { FusP f; f.c13 = rdLin(c); f.c31 = rdLin(c); f.ffa = rdLin(c);
                            f.ffb = rdLin(c); f.mca = rdLin(c); f.mcb = rdLin(c); return f; }
static DecP rdDec(Cur& c) { DecP d; d.atta = rdLin(c); d.attb = rdLin(c);
                            d.gat = rdGat(c); d.gct = rdGct(c); return d; }
static EncP rdEnc(Cur& c) { EncP e; e.atta = rdLin(c); e.attb = rdLin(c); e.fus = rdFus(c);
                            e.gat = rdGat(c); e.gct = rdGct(c); e.pool = rdLin(c); return e; }

struct Bump {
  char* base; size_t off;
  void* get(size_t bytes) { void* r = base + off; off += (bytes + 255) & ~(size_t)255; return r; }
  float*    f(size_t n)   { return (float*)get(n * sizeof(float)); }
  int*      i32(size_t n) { return (int*)get(n * sizeof(int)); }
  unsigned* u32(size_t n) { return (unsigned*)get(n * sizeof(unsigned)); }
};

static inline unsigned gblocks(size_t n) {
  size_t b = (n + 255) / 256;
  if (b > (1u << 20)) b = 1u << 20;
  if (b == 0) b = 1;
  return (unsigned)b;
}

// GEMM wrapper: packs W to bf16 fragment layout (scratch), then launches the
// WMMA kernel.
static void gemm(hipStream_t st, Bump& scr, const float* A, int lda,
                 const float* W, int ldw, const float* b,
                 float* C, int ldc, int M, int N, int K, int act) {
  int nK = K / 32, nN = N / 16;
  __bf16* wpk = (__bf16*)scr.get((size_t)nK * nN * 512 * sizeof(__bf16));
  k_pack_w<<<gblocks((size_t)nK * nN * 512), 256, 0, st>>>(W, ldw, wpk, nK, nN);
  dim3 g((M + 127) / 128, nN);
  k_gemm<<<g, 256, 0, st>>>(A, lda, wpk, b, C, ldc, M, K, act);
}

// gat_conv: z = x@w; e = leaky(el[src]+er[dst]); dst-softmax; agg; (+res); elu
static float* run_gat_conv(hipStream_t st, Bump& scr, const float* x, int din, int Hd,
                           const float* w, const float* al, const float* ar,
                           const float* wres, const int* src, const int* dst, int n) {
  int Ds = HEADS * Hd;
  float* z = scr.f((size_t)n * Ds);
  gemm(st, scr, x, din, w, Ds, nullptr, z, Ds, n, Ds, din, 0);
  float* el = scr.f((size_t)n * 4);
  float* er = scr.f((size_t)n * 4);
  k_attn_coef<<<gblocks((size_t)n * 4), 256, 0, st>>>(z, al, ar, el, er, n, Hd);
  unsigned* mm = scr.u32((size_t)(n + 1) * 4);
  k_fill_u32<<<gblocks((size_t)(n + 1) * 4), 256, 0, st>>>(mm, 0x007FFFFFu, (size_t)(n + 1) * 4); // mono(-inf)
  float* eb = scr.f((size_t)EDGES * 4);
  k_edge_e_max<<<gblocks((size_t)EDGES * 4), 256, 0, st>>>(src, dst, el, er, eb, mm, (size_t)EDGES * 4);
  float* mf = scr.f((size_t)(n + 1) * 4);
  k_mono_fix<<<gblocks((size_t)(n + 1) * 4), 256, 0, st>>>(mm, mf, (size_t)(n + 1) * 4);
  float* ss = scr.f((size_t)(n + 1) * 4);
  k_fill_f32<<<gblocks((size_t)(n + 1) * 4), 256, 0, st>>>(ss, 0.f, (size_t)(n + 1) * 4);
  float* pb = scr.f((size_t)EDGES * 4);
  k_edge_p_sum<<<gblocks((size_t)EDGES * 4), 256, 0, st>>>(dst, eb, mf, pb, ss, (size_t)EDGES * 4);
  float* agg = scr.f((size_t)(n + 1) * Ds);
  k_fill_f32<<<gblocks((size_t)(n + 1) * Ds), 256, 0, st>>>(agg, 0.f, (size_t)(n + 1) * Ds);
  k_edge_agg<<<dim3(EDGES), 256, 0, st>>>(src, dst, pb, ss, z, agg, Hd);
  float* out = scr.f((size_t)n * Ds);
  if (wres) {
    float* res = scr.f((size_t)n * Ds);
    gemm(st, scr, x, din, wres, Ds, nullptr, res, Ds, n, Ds, din, 0);
    k_add_elu<<<gblocks((size_t)n * Ds), 256, 0, st>>>(agg, res, out, (size_t)n * Ds);
  } else {
    k_elu<<<gblocks((size_t)n * Ds), 256, 0, st>>>(agg, out, (size_t)n * Ds);
  }
  return out;
}

static float* run_gat_block(hipStream_t st, Bump& scr, const Gat& G, const float* x,
                            int din, int Hd, const int* src, const int* dst, int n) {
  float* h1 = run_gat_conv(st, scr, x, din, Hd, G.w0, G.al0, G.ar0, nullptr, src, dst, n);
  return run_gat_conv(st, scr, h1, HEADS * Hd, Hd, G.w1, G.al1, G.ar1, G.wres, src, dst, n);
}

static float* run_gct(hipStream_t st, Bump& scr, const Gct& G, const float* x,
                      int din, int ghid, int C, int n) {
  float* t1 = scr.f((size_t)n * ghid);
  gemm(st, scr, x, din, G.m1a.w, ghid, G.m1a.b, t1, ghid, n, ghid, din, 1);
  float* h = scr.f((size_t)n * C);
  gemm(st, scr, t1, ghid, G.m1b.w, C, G.m1b.b, h, C, n, C, ghid, 0);
  float* gh = scr.f((size_t)n * C);
  k_gct_gate<<<dim3((n + 7) / 8), 256, 0, st>>>(h, gh, n, C);
  float* t2 = scr.f((size_t)n * ghid);
  gemm(st, scr, gh, C, G.m2a.w, ghid, G.m2a.b, t2, ghid, n, ghid, C, 1);
  float* c = scr.f((size_t)n * C);
  gemm(st, scr, t2, ghid, G.m2b.w, C, G.m2b.b, c, C, n, C, ghid, 0);
  return c;
}

static void run_fusion(hipStream_t st, Bump& scr, const FusP& F, const float* s,
                       const float* c, int Ds, int C, int ff, int n, float* fused) {
  float* cp1 = scr.f((size_t)n * Ds);
  gemm(st, scr, c, C, F.mca.w, Ds, F.mca.b, cp1, Ds, n, Ds, C, 1);
  float* cp = scr.f((size_t)n * Ds);
  gemm(st, scr, cp1, Ds, F.mcb.w, Ds, F.mcb.b, cp, Ds, n, Ds, Ds, 0);
  int D2 = 2 * Ds;
  float* a = scr.f((size_t)n * D2);
  float* b = scr.f((size_t)n * D2);
  gemm(st, scr, s, Ds, F.c13.w, Ds, F.c13.b, a, D2, n, Ds, Ds, 0);
  gemm(st, scr, s, Ds, F.c31.w, Ds, F.c31.b, b, D2, n, Ds, Ds, 0);
  k_copy_cols<<<gblocks((size_t)n * Ds), 256, 0, st>>>(cp, Ds, a + Ds, D2, n, Ds);
  k_copy_cols<<<gblocks((size_t)n * Ds), 256, 0, st>>>(cp, Ds, b + Ds, D2, n, Ds);
  float* ab = scr.f((size_t)n * D2);
  k_mul<<<gblocks((size_t)n * D2), 256, 0, st>>>(a, b, ab, (size_t)n * D2);
  float* t = scr.f((size_t)n * ff);
  gemm(st, scr, ab, D2, F.ffa.w, ff, F.ffa.b, t, ff, n, ff, D2, 1);
  gemm(st, scr, t, ff, F.ffb.w, ff, F.ffb.b, fused, ff, n, ff, ff, 0);
}

static void run_topk(hipStream_t st, Bump& scr, const float* scores, int n, int K, int* idx) {
  TopkState* stt = (TopkState*)scr.get(sizeof(TopkState));
  k_topk_init<<<1, 256, 0, st>>>(stt, (unsigned)K);
  for (int shift = 24; shift >= 0; shift -= 8) {
    k_topk_hist<<<gblocks(n), 256, 0, st>>>(scores, n, stt, shift);
    k_topk_scan<<<1, 256, 0, st>>>(stt, shift);
  }
  k_topk_gt<<<gblocks(n), 256, 0, st>>>(scores, n, stt, idx);
  k_topk_eq<<<gblocks(n), 256, 0, st>>>(scores, n, stt, idx, (unsigned)K);
}

static void run_encoder(hipStream_t st, Bump scr, const EncP& P, const float* x,
                        const int* src, const int* dst, int n, int Kk,
                        int din, int Hd, int ghid, int C, int ff, int atth,
                        float* xnext, float* fused, int* idx, int* s2, int* d2) {
  int Ds = HEADS * Hd, dcon = Ds + C;
  float* s = run_gat_block(st, scr, P.gat, x, din, Hd, src, dst, n);
  float* c = run_gct(st, scr, P.gct, x, din, ghid, C, n);
  run_fusion(st, scr, P.fus, s, c, Ds, C, ff, n, fused);
  float* con = scr.f((size_t)n * dcon);
  k_copy_cols<<<gblocks((size_t)n * Ds), 256, 0, st>>>(s, Ds, con, dcon, n, Ds);
  k_copy_cols<<<gblocks((size_t)n * C), 256, 0, st>>>(c, C, con + Ds, dcon, n, C);
  float* t = scr.f((size_t)n * atth);
  gemm(st, scr, con, dcon, P.atta.w, atth, P.atta.b, t, atth, n, atth, dcon, 1);
  float* w2 = scr.f((size_t)n * dcon);
  gemm(st, scr, t, atth, P.attb.w, dcon, P.attb.b, w2, dcon, n, dcon, atth, 0);
  float* att = scr.f((size_t)n * dcon);
  k_mul<<<gblocks((size_t)n * dcon), 256, 0, st>>>(con, w2, att, (size_t)n * dcon);
  float* scores = scr.f(n);
  k_rowdot<<<dim3((n + 7) / 8), 256, 0, st>>>(att, P.pool.w, P.pool.b, scores, n, dcon);
  run_topk(st, scr, scores, n, Kk, idx);
  k_gather_rows<<<dim3(Kk), 256, 0, st>>>(att, idx, xnext, dcon);
  if (s2) {
    int* map = scr.i32((size_t)n + 1);
    k_fill_i32<<<gblocks((size_t)n + 1), 256, 0, st>>>(map, -1, (size_t)n + 1);
    k_set_map<<<gblocks(Kk), 256, 0, st>>>(idx, map, Kk);
    k_remap<<<gblocks(EDGES), 256, 0, st>>>(src, dst, map, s2, d2, Kk, (size_t)EDGES);
  }
}

static void run_decoder(hipStream_t st, Bump scr, const DecP& P, const float* xin,
                        const int* src, const int* dst, int n_out, const int* idx,
                        int kin, const float* fused, int din, int Hd, int ghid,
                        int C, int atth, float* hout) {
  int Ds = HEADS * Hd, dcon = Ds + C;
  float* up = scr.f((size_t)n_out * din);
  k_copy_cols<<<gblocks((size_t)n_out * din), 256, 0, st>>>(fused, din, up, din, n_out, din);
  k_scatter_add<<<dim3(kin), 256, 0, st>>>(xin, idx, up, din);
  float* s = run_gat_block(st, scr, P.gat, up, din, Hd, src, dst, n_out);
  float* c = run_gct(st, scr, P.gct, up, din, ghid, C, n_out);
  float* con = scr.f((size_t)n_out * dcon);
  k_copy_cols<<<gblocks((size_t)n_out * Ds), 256, 0, st>>>(s, Ds, con, dcon, n_out, Ds);
  k_copy_cols<<<gblocks((size_t)n_out * C), 256, 0, st>>>(c, C, con + Ds, dcon, n_out, C);
  float* t = scr.f((size_t)n_out * atth);
  gemm(st, scr, con, dcon, P.atta.w, atth, P.atta.b, t, atth, n_out, atth, dcon, 1);
  float* w2 = scr.f((size_t)n_out * dcon);
  gemm(st, scr, t, atth, P.attb.w, dcon, P.attb.b, w2, dcon, n_out, dcon, atth, 0);
  k_mul<<<gblocks((size_t)n_out * dcon), 256, 0, st>>>(con, w2, hout, (size_t)n_out * dcon);
}

// ============================================================================

extern "C" void kernel_launch(void* const* d_in, const int* in_sizes, int n_in,
                              void* d_out, int out_size, void* d_ws, size_t ws_size,
                              hipStream_t stream) {
  (void)in_sizes; (void)n_in; (void)out_size; (void)ws_size;

  Cur c{d_in, 0};
  const int* dstE = (const int*)d_in[c.i++];
  DecP dec1 = rdDec(c), dec2 = rdDec(c), dec3 = rdDec(c);
  EncP enc1 = rdEnc(c), enc2 = rdEnc(c), enc3 = rdEnc(c);
  Lin gcn1 = rdLin(c), gcn2 = rdLin(c);
  const int* srcE = (const int*)d_in[c.i++];
  const float* x  = (const float*)d_in[c.i++];

  // Persistent workspace (carried across stages), then a scratch arena that
  // is reset at every stage boundary (Bump passed by value).
  Bump pers{(char*)d_ws, 0};
  float* x1  = pers.f((size_t)KK1 * 256);
  float* x2  = pers.f((size_t)KK2 * 512);
  float* x3  = pers.f((size_t)KK3 * 1024);
  float* f1  = pers.f((size_t)N0  * 256);
  float* f2  = pers.f((size_t)KK1 * 512);
  float* f3  = pers.f((size_t)KK2 * 1024);
  float* hd3 = pers.f((size_t)KK2 * 512);
  float* hd2 = pers.f((size_t)KK1 * 256);
  float* hd1 = pers.f((size_t)N0  * 256);
  int* idx1 = pers.i32(KK1);
  int* idx2 = pers.i32(KK2);
  int* idx3 = pers.i32(KK3);
  int* s1 = pers.i32(EDGES); int* d1 = pers.i32(EDGES);
  int* s2 = pers.i32(EDGES); int* d2 = pers.i32(EDGES);
  Bump scr{pers.base + pers.off, 0};

  // Encoders
  run_encoder(stream, scr, enc1, x,  srcE, dstE, N0,  KK1, 256, 48, 128, 64,  256,  256,  x1, f1, idx1, s1, d1);
  run_encoder(stream, scr, enc2, x1, s1,   d1,   KK1, KK2, 256, 96, 256, 128, 512,  512,  x2, f2, idx2, s2, d2);
  run_encoder(stream, scr, enc3, x2, s2,   d2,   KK2, KK3, 512, 192, 512, 256, 1024, 1024, x3, f3, idx3, nullptr, nullptr);

  // Decoders
  run_decoder(stream, scr, dec3, x3,  s2,   d2,   KK2, idx3, KK3, f3, 1024, 96, 256, 128, 512, hd3);
  run_decoder(stream, scr, dec2, hd3, s1,   d1,   KK1, idx2, KK2, f2, 512,  48, 128, 64,  256, hd2);
  run_decoder(stream, scr, dec1, hd2, srcE, dstE, N0,  idx1, KK1, f1, 256,  48, 128, 64,  256, hd1);

  // GCN head (GraphConv norm='both', 2 layers) -> d_out [N0, 16]
  Bump g = scr;
  float* ns = g.f(N0);
  float* nd = g.f(N0);
  k_fill_f32<<<gblocks(N0), 256, 0, stream>>>(ns, 0.f, N0);
  k_fill_f32<<<gblocks(N0), 256, 0, stream>>>(nd, 0.f, N0);
  k_degree<<<gblocks(EDGES), 256, 0, stream>>>(srcE, dstE, ns, nd, (size_t)EDGES);
  k_invsqrt<<<gblocks(N0), 256, 0, stream>>>(ns, N0);
  k_invsqrt<<<gblocks(N0), 256, 0, stream>>>(nd, N0);
  float* agg = g.f((size_t)N0 * 256);
  k_fill_f32<<<gblocks((size_t)N0 * 256), 256, 0, stream>>>(agg, 0.f, (size_t)N0 * 256);
  k_gcn_agg<<<dim3(EDGES), 256, 0, stream>>>(srcE, dstE, hd1, ns, agg, 256);
  k_scale_rows<<<gblocks((size_t)N0 * 256), 256, 0, stream>>>(agg, nd, N0, 256);
  float* t = g.f((size_t)N0 * 256);
  gemm(stream, g, agg, 256, gcn1.w, 256, gcn1.b, t, 256, N0, 256, 256, 1);
  float* agg2 = g.f((size_t)N0 * 256);
  k_fill_f32<<<gblocks((size_t)N0 * 256), 256, 0, stream>>>(agg2, 0.f, (size_t)N0 * 256);
  k_gcn_agg<<<dim3(EDGES), 256, 0, stream>>>(srcE, dstE, t, ns, agg2, 256);
  k_scale_rows<<<gblocks((size_t)N0 * 256), 256, 0, stream>>>(agg2, nd, N0, 256);
  gemm(stream, g, agg2, 256, gcn2.w, 16, gcn2.b, (float*)d_out, 16, N0, 16, 256, 0);
}
